// MAGNN_mptype_layer_12549894439028
// MI455X (gfx1250) — matrix operand order repaired and checked by reading, not compiled, not run on hardware
//
#include <hip/hip_runtime.h>

// ---------------------------------------------------------------------------
// MAGNN mptype layer for MI455X (gfx1250, wave32).
// Sizes from reference: N=50000, E=200000, L=4, ODIM=64, NHEAD=8, NREL=3.
//
// Passes:
//   0a) init_out   : out[0..N) = (float)iftargets[:,0]; out[N..N+N*512) = 0
//   0b) init_stats : amax (monotone-uint -inf) = 0, asum = 0
//   1)  edge kernel: TDM-stage weights to LDS (overlapped with gather);
//                    gather+rotate+mean -> hidden; WMMA f16 logits -> a;
//                    atomicMax segment max
//   2)  softmax    : aexp = exp(a - amax[dst]); atomicAdd asum
//   3)  scatter    : ft[dst,h,:] += (aexp/asum) * hidden   (atomic f32 adds)
//
// Workspace (floats): hidden E*64 | a/aexp E*8 | amax N*8 (uint) | asum N*8
//   = 51.2MB + 6.4MB + 1.6MB + 1.6MB ~= 61MB
// ---------------------------------------------------------------------------

typedef __attribute__((ext_vector_type(16))) _Float16 v16h;
typedef __attribute__((ext_vector_type(8)))  _Float16 v8h;
typedef __attribute__((ext_vector_type(8)))  float    v8f;
typedef __attribute__((ext_vector_type(4)))  unsigned v4u;
typedef __attribute__((ext_vector_type(8)))  int      v8i;
typedef __attribute__((ext_vector_type(4)))  int      v4i;

#define ODIM   64
#define LPATH  4
#define NHEAD  8
#define ALPHA  0.01f
#define SA_PITCH 136   // halves per A-tile row (128 + 8 pad: kills LDS bank conflicts)

#if __has_builtin(__builtin_amdgcn_tensor_load_to_lds)
#define HAVE_TDM 1
#else
#define HAVE_TDM 0
#endif

// Monotone mapping float -> uint so unsigned atomicMax == float max.
__device__ __forceinline__ unsigned mono_f32(float f) {
  unsigned u = __float_as_uint(f);
  return (u & 0x80000000u) ? ~u : (u | 0x80000000u);
}
__device__ __forceinline__ float demono_f32(unsigned m) {
  unsigned u = (m & 0x80000000u) ? (m & 0x7FFFFFFFu) : ~m;
  return __uint_as_float(u);
}

#if HAVE_TDM
// Issue one TDM descriptor: contiguous 1-D tile of `nelem` f32 from global
// to LDS byte address `lds_byte`. D# packing per CDNA5 ISA 08_async_tensor
// sections 8.3/8.4: group0 = {count|flags, lds_addr, ga_lo, ga_hi|type=2};
// group1 = {data_size, tensor_dim0, tensor_dim1=1, tile_dim0=nelem,
//           tile_dim1=1, stride0=nelem}. tile == tensor -> no OOB handling.
// This toolchain (clang-23 / therock-10.0) exposes the 6-arg builtin:
//   (uint32x4 g0, int32x8 g1, int32x4 g2, int32x4 g3, int32x8 extra, i32 cpol)
__device__ __forceinline__ void tdm_load_1d_f32(unsigned lds_byte,
                                                const float* gsrc,
                                                unsigned nelem) {
  unsigned long long ga = (unsigned long long)(size_t)gsrc;
  v4u g0;
  g0[0] = 1u;                                            // count=1 (valid user D#)
  g0[1] = lds_byte;                                      // lds_addr
  g0[2] = (unsigned)(ga & 0xFFFFFFFFull);                // global_addr[31:0]
  g0[3] = (unsigned)((ga >> 32) & 0x01FFFFFFull)         // global_addr[56:32]
        | (2u << 30);                                    // type = 2 ("image")
  v8i g1;
  g1[0] = (int)(2u << 16);                               // data_size = 4B
  g1[1] = (int)((nelem & 0xFFFFu) << 16);                // tensor_dim0[15:0]
  g1[2] = (int)(((nelem >> 16) & 0xFFFFu) | (1u << 16)); // dim0[31:16] | tensor_dim1=1
  g1[3] = (int)((nelem & 0xFFFFu) << 16);                // tile_dim0 = nelem
  g1[4] = (int)1;                                        // tile_dim1 = 1
  g1[5] = (int)nelem;                                    // tensor_dim0_stride lo
  g1[6] = 0;                                             // stride0 hi | stride1 lo
  g1[7] = 0;
  v4i z4 = {0, 0, 0, 0};
  v8i z8 = {0, 0, 0, 0, 0, 0, 0, 0};
  __builtin_amdgcn_tensor_load_to_lds(g0, g1, z4, z4, z8, 0);
}
#endif

// ---------------------------------------------------------------------------
__global__ void magnn_init_out_kernel(const int* __restrict__ ift,
                                      float* __restrict__ out, int N) {
  long long i = (long long)blockIdx.x * blockDim.x + threadIdx.x;
  long long total = (long long)N * (1 + NHEAD * ODIM);
  if (i >= total) return;
  out[i] = (i < N) ? (float)ift[2 * i] : 0.0f;   // first output + zeroed ft
}

__global__ void magnn_init_stats_kernel(unsigned* __restrict__ amax,
                                        float* __restrict__ asum, int N) {
  int i = blockIdx.x * blockDim.x + threadIdx.x;
  if (i >= N * NHEAD) return;
  amax[i] = 0u;      // mono encoding below any finite float
  asum[i] = 0.0f;
}

// ---------------------------------------------------------------------------
// Pass 1: 128 threads = 4 waves, 64 edges per block.
//   - wave 0 TDM-stages Wcat (4KB) into LDS; DMA overlaps with phase 1
//   - phase 1: thread (el, seg) gathers 32 features of edge el, rotates, means
//   - phase 2: wave w does a 16x16x128 WMMA tile (4 x v_wmma_f32_16x16x32_f16)
__global__ __launch_bounds__(128)
void magnn_edge_kernel(const int*   __restrict__ mp,
                       const float* __restrict__ nf,
                       const float* __restrict__ rv,
                       const float* __restrict__ w1,
                       const float* __restrict__ w2,
                       float*       __restrict__ ws_hidden,
                       float*       __restrict__ ws_a,
                       unsigned*    __restrict__ ws_amax,
                       int E)
{
  __shared__ _Float16 sA[64 * SA_PITCH];   // 64 edges x 128 halves (+pad)
  __shared__ float    sW[1024];            // [0..511]=attn1_w, [512..1023]=attn2
  const int t  = threadIdx.x;
  const int e0 = blockIdx.x * 64;

#if HAVE_TDM
  // Kick off the weight DMA first; it completes while phase 1 gathers.
  if (t < 32) {
    unsigned wbase = (unsigned)(size_t)(void*)sW;   // flat addr low 32 = LDS byte offset
    tdm_load_1d_f32(wbase,        w1, NHEAD * ODIM);
    tdm_load_1d_f32(wbase + 2048, w2, NHEAD * ODIM);
  }
#else
  // Fallback: cooperative global->LDS copy of the weights.
  for (int i = t; i < 1024; i += 128)
    sW[i] = (i < 512) ? w1[i] : w2[i - 512];
#endif

  // ----------------------- Phase 1: gather + rotate ------------------------
  {
    const int el  = t >> 1;                 // edge within block (0..63)
    const int seg = t & 1;                  // feature half: [seg*32, seg*32+32)
    const int egr = e0 + el;
    const int eg  = egr < E ? egr : E - 1;  // clamp (WMMA needs full EXEC)
    const int i0 = mp[eg*4+0], i1 = mp[eg*4+1], i2 = mp[eg*4+2], i3 = mp[eg*4+3];
    __builtin_prefetch(nf + (size_t)i2 * ODIM + seg * 32, 0, 0);
    __builtin_prefetch(nf + (size_t)i1 * ODIM + seg * 32, 0, 0);
    __builtin_prefetch(nf + (size_t)i0 * ODIM + seg * 32, 0, 0);

    float x[32], hid[32], fre[16], fim[16];

    // level 3: rotation = identity; this row is also "center" (A-tile K 0..63)
    {
      const float4* p = (const float4*)(nf + (size_t)i3 * ODIM + seg * 32);
      #pragma unroll
      for (int j = 0; j < 8; ++j) {
        float4 v = p[j];
        x[4*j] = v.x; x[4*j+1] = v.y; x[4*j+2] = v.z; x[4*j+3] = v.w;
      }
      #pragma unroll
      for (int k = 0; k < 32; ++k) {
        hid[k] = x[k];
        sA[el * SA_PITCH + seg * 32 + k] = (_Float16)x[k];
      }
    }

    // levels 2,1,0: f3=1, f2=r2, f1=f2*r1, f0=f1*r0 (ETYPES=(0,2,4) -> rv[2],rv[1],rv[0])
    const int idxs[3] = { i2, i1, i0 };
    const int rels[3] = { 2, 1, 0 };
    #pragma unroll
    for (int s = 0; s < 3; ++s) {
      const float* rp = rv + rels[s] * ODIM + seg * 32;
      #pragma unroll
      for (int j = 0; j < 16; ++j) {
        float rr = rp[2*j], ri = rp[2*j+1];
        float inv = rsqrtf(rr * rr + ri * ri);   // per-element complex normalize
        rr *= inv; ri *= inv;
        if (s == 0) { fre[j] = rr; fim[j] = ri; }
        else {
          float nr = fre[j]*rr - fim[j]*ri;
          float ni = fre[j]*ri + fim[j]*rr;
          fre[j] = nr; fim[j] = ni;
        }
      }
      const float4* p = (const float4*)(nf + (size_t)idxs[s] * ODIM + seg * 32);
      #pragma unroll
      for (int j = 0; j < 8; ++j) {
        float4 v = p[j];
        x[4*j] = v.x; x[4*j+1] = v.y; x[4*j+2] = v.z; x[4*j+3] = v.w;
      }
      #pragma unroll
      for (int j = 0; j < 16; ++j) {
        hid[2*j]   += x[2*j]*fre[j] - x[2*j+1]*fim[j];
        hid[2*j+1] += x[2*j]*fim[j] + x[2*j+1]*fre[j];
      }
    }

    #pragma unroll
    for (int k = 0; k < 32; ++k) {
      hid[k] *= 0.25f;                                   // mean over L=4
      sA[el * SA_PITCH + 64 + seg * 32 + k] = (_Float16)hid[k]; // A-tile K 64..127
    }
    if (egr < E) {
      float4* ho = (float4*)(ws_hidden + (size_t)egr * ODIM + seg * 32);
      #pragma unroll
      for (int j = 0; j < 8; ++j) {
        float4 v; v.x = hid[4*j]; v.y = hid[4*j+1]; v.z = hid[4*j+2]; v.w = hid[4*j+3];
        ho[j] = v;
      }
    }
  }

#if HAVE_TDM
  if (t < 32) __builtin_amdgcn_s_wait_tensorcnt(0);  // weights landed in LDS
#endif
  __syncthreads();

  // ------------------- Phase 2: WMMA attention logits ----------------------
  {
    const int wave = t >> 5;          // tile id, 16 edges each
    const int lane = t & 31;
    const int n    = lane & 15;       // B/D column = head (valid < 8)
    const bool hi  = lane >= 16;

    // B fragments from LDS-staged Wcat (128 x 16); col n<8 = [W1_row_n ; W2_row_n].
    // Layout (16-bit B 32x16, wave32): lanes 0-15 col=lane K=0..7,16..23; lanes 16-31 K=8..15,24..31.
    v16h bfrag[4];
    #pragma unroll
    for (int c = 0; c < 4; ++c) {
      v16h b;
      #pragma unroll
      for (int j = 0; j < 16; ++j) b[j] = (_Float16)0.0f;
      if (n < NHEAD) {
        const int kb = hi ? 8 : 0;
        #pragma unroll
        for (int j = 0; j < 16; ++j) {
          int K = 32 * c + kb + (j < 8 ? j : j + 8);
          float w = (K < ODIM) ? sW[n * ODIM + K] : sW[512 + n * ODIM + (K - ODIM)];
          b[j] = (_Float16)w;
        }
      }
      bfrag[c] = b;
    }

    // A fragments from LDS (all loaded up front so the 8 ds_load_b128 pipeline):
    //   lanes 0-15 : M=lane,  elems 0..7 = K 0..7,  elems 8..15 = K 16..23
    //   lanes 16-31: M=ln-16, elems 0..7 = K 8..15, elems 8..15 = K 24..31
    const int row   = wave * 16 + n;     // edge-in-block (M = lane&15)
    const int kbase = hi ? 8 : 0;
    v16h afrag[4];
    #pragma unroll
    for (int c = 0; c < 4; ++c) {
      const v8h* lo  = (const v8h*)&sA[row * SA_PITCH + 32*c + kbase];       // 16B aligned
      const v8h* hiq = (const v8h*)&sA[row * SA_PITCH + 32*c + 16 + kbase];  // 16B aligned
      v8h alo = *lo, ahi = *hiq;
      v16h a;
      #pragma unroll
      for (int j = 0; j < 8; ++j) { a[j] = alo[j]; a[8 + j] = ahi[j]; }
      afrag[c] = a;
    }

    v8f acc = {};
    #pragma unroll
    for (int c = 0; c < 4; ++c)   // 4x v_wmma_f32_16x16x32_f16, back-to-back
      acc = __builtin_amdgcn_wmma_f32_16x16x32_f16(false, afrag[c], false, bfrag[c],
                                                   (short)0, acc, false, false);

    // leaky_relu on the 16x16 accumulator
    #pragma unroll
    for (int v = 0; v < 8; ++v) { float y = acc[v]; acc[v] = y > 0.0f ? y : ALPHA * y; }

    // C/D layout: acc[v] holds M = v + (hi?8:0), col N = lane&15
    if (n < NHEAD) {
      #pragma unroll
      for (int v = 0; v < 8; ++v) {
        int m  = wave * 16 + v + (hi ? 8 : 0);
        int eg = e0 + m;
        if (eg < E) {
          ws_a[(size_t)eg * NHEAD + n] = acc[v];
          int dst = mp[eg * 4 + 3];
          atomicMax(&ws_amax[(size_t)dst * NHEAD + n], mono_f32(acc[v]));
        }
      }
    }
  }
}

// ---------------------------------------------------------------------------
// Pass 2: segment softmax numerator + denominator.
__global__ void magnn_softmax_kernel(const int* __restrict__ mp,
                                     float* __restrict__ ws_a,
                                     const unsigned* __restrict__ ws_amax,
                                     float* __restrict__ ws_asum, int E) {
  int i = blockIdx.x * blockDim.x + threadIdx.x;
  if (i >= E * NHEAD) return;
  int e = i >> 3, h = i & 7;
  int dst = mp[e * 4 + 3];
  float mx = demono_f32(ws_amax[(size_t)dst * NHEAD + h]);
  float ex = __expf(ws_a[i] - mx);
  ws_a[i] = ex;                                   // in-place: a -> aexp
  atomicAdd(&ws_asum[(size_t)dst * NHEAD + h], ex);
}

// ---------------------------------------------------------------------------
// Pass 3: ft[dst,h,c] += (aexp/asum) * hidden[e,c].  64 threads per edge.
__global__ __launch_bounds__(256)
void magnn_scatter_kernel(const int* __restrict__ mp,
                          const float* __restrict__ ws_hidden,
                          const float* __restrict__ ws_a,
                          const float* __restrict__ ws_asum,
                          float* __restrict__ out_ft, int E) {
  int e = blockIdx.x * 4 + (threadIdx.x >> 6);
  int c = threadIdx.x & 63;
  if (e >= E) return;
  int dst = mp[e * 4 + 3];
  float hv = ws_hidden[(size_t)e * ODIM + c];
  const float* sums = ws_asum + (size_t)dst * NHEAD;
  const float* exps = ws_a + (size_t)e * NHEAD;
  float* obase = out_ft + (size_t)dst * NHEAD * ODIM + c;
  #pragma unroll
  for (int h = 0; h < NHEAD; ++h) {
    float att = exps[h] / sums[h];
    atomicAdd(obase + h * ODIM, att * hv);
  }
}

// ---------------------------------------------------------------------------
extern "C" void kernel_launch(void* const* d_in, const int* in_sizes, int n_in,
                              void* d_out, int out_size, void* d_ws, size_t ws_size,
                              hipStream_t stream) {
  (void)n_in; (void)out_size; (void)ws_size;
  const int*   mp  = (const int*)d_in[0];     // (E, 4) int32
  const int*   ift = (const int*)d_in[1];     // (N, 2) int32
  const float* nf  = (const float*)d_in[2];   // (N, 64) f32
  const float* rv  = (const float*)d_in[3];   // (3, 32, 2) f32
  const float* w1  = (const float*)d_in[4];   // (8, 64) f32
  const float* w2  = (const float*)d_in[5];   // (1, 8, 64) f32

  const int E = in_sizes[0] / LPATH;
  const int N = in_sizes[2] / ODIM;

  float*    ws_hidden = (float*)d_ws;                          // E*64
  float*    ws_a      = ws_hidden + (size_t)E * ODIM;          // E*8
  unsigned* ws_amax   = (unsigned*)(ws_a + (size_t)E * NHEAD); // N*8
  float*    ws_asum   = (float*)(ws_amax + (size_t)N * NHEAD); // N*8

  float* out    = (float*)d_out;
  float* out_ft = out + N;

  long long tot_out = (long long)N * (1 + NHEAD * ODIM);
  magnn_init_out_kernel<<<(unsigned)((tot_out + 255) / 256), 256, 0, stream>>>(ift, out, N);
  magnn_init_stats_kernel<<<(N * NHEAD + 255) / 256, 256, 0, stream>>>(ws_amax, ws_asum, N);
  magnn_edge_kernel<<<(E + 63) / 64, 128, 0, stream>>>(mp, nf, rv, w1, w2,
                                                       ws_hidden, ws_a, ws_amax, E);
  magnn_softmax_kernel<<<(E * NHEAD + 255) / 256, 256, 0, stream>>>(mp, ws_a, ws_amax,
                                                                    ws_asum, E);
  magnn_scatter_kernel<<<(E + 3) / 4, 256, 0, stream>>>(mp, ws_hidden, ws_a,
                                                        ws_asum, out_ft, E);
}
